// AttentionBlock_79559974191215
// MI455X (gfx1250) — matrix operand order repaired and checked by reading, not compiled, not run on hardware
//
#include <hip/hip_runtime.h>

typedef __attribute__((ext_vector_type(16))) _Float16 v16h;
typedef __attribute__((ext_vector_type(8)))  float    v8f;

#define WMMA_F16(a, b, c) \
  __builtin_amdgcn_wmma_f32_16x16x32_f16(false, (a), false, (b), (short)0, (c), false, false)

// ---------------------------------------------------------------------------
// WMMA fragment loaders (wave32, 16x16x32 f16)
// A matrix 16x32 (MxK): lane m = lane&15; halves 0..7 -> K = (lane>>4)*8 + j,
// halves 8..15 -> K = 16 + (lane>>4)*8 + (j-8).  Two 16B contiguous loads.
// ---------------------------------------------------------------------------
__device__ __forceinline__ v16h load_frag_a(const _Float16* __restrict__ base, int ld) {
  const int lane = threadIdx.x & 31;
  const int row  = lane & 15;
  const int k0   = (lane >> 4) << 3;  // 0 or 8
  const _Float16* p = base + (size_t)row * ld + k0;
  v16h f;
#pragma unroll
  for (int i = 0; i < 8; ++i) { f[i] = p[i]; f[i + 8] = p[i + 16]; }
  return f;
}

// B matrix 32x16 (KxN) given storage element [n][k] (i.e. B^T rows contiguous):
// lane n = lane&15; halves 0..7 -> K = (lane>>4)*8 + j, halves 8..15 -> K = 16 + (lane>>4)*8 + (j-8).
__device__ __forceinline__ v16h load_frag_b(const _Float16* __restrict__ base, int ld) {
  const int lane = threadIdx.x & 31;
  const _Float16* p = base + (size_t)(lane & 15) * ld + ((lane >> 4) << 3);
  v16h f;
#pragma unroll
  for (int i = 0; i < 8; ++i) { f[i] = p[i]; f[i + 8] = p[i + 16]; }
  return f;
}

// C/D 16x16 f32: lane col n = lane&15; VGPR r -> row m = r + 8*(lane>>4).

// ---------------------------------------------------------------------------
// GroupNorm: 32 groups of 16 channels, reduce over 4096 spatial * 16 ch.
// One workgroup (256 threads) per (batch, group). Writes hn in f16.
// ---------------------------------------------------------------------------
__global__ void groupnorm_kernel(const float* __restrict__ x,
                                 const float* __restrict__ gamma,
                                 const float* __restrict__ beta,
                                 _Float16* __restrict__ hn) {
  const int bg = blockIdx.x;          // 0..63
  const int b  = bg >> 5;
  const int g  = bg & 31;
  const int t  = threadIdx.x;         // 256 threads
  const float* xb = x + (size_t)b * 4096 * 512;

  float s = 0.f, ss = 0.f;
  for (int sp = t; sp < 4096; sp += 256) {
    const float* p = xb + (size_t)sp * 512 + g * 16;
#pragma unroll
    for (int cc = 0; cc < 16; ++cc) { float v = p[cc]; s += v; ss += v * v; }
  }
  __shared__ float rs[256], rss[256];
  rs[t] = s; rss[t] = ss;
  __syncthreads();
  for (int off = 128; off > 0; off >>= 1) {
    if (t < off) { rs[t] += rs[t + off]; rss[t] += rss[t + off]; }
    __syncthreads();
  }
  const float inv  = 1.0f / 65536.0f;
  const float mean = rs[0] * inv;
  const float var  = rss[0] * inv - mean * mean;
  const float rstd = rsqrtf(var + 1e-5f);

  _Float16* hb = hn + (size_t)b * 4096 * 512;
  for (int sp = t; sp < 4096; sp += 256) {
    const float* p = xb + (size_t)sp * 512 + g * 16;
    _Float16*   q = hb + (size_t)sp * 512 + g * 16;
#pragma unroll
    for (int cc = 0; cc < 16; ++cc) {
      const int ch = g * 16 + cc;
      q[cc] = (_Float16)((p[cc] - mean) * rstd * gamma[ch] + beta[ch]);
    }
  }
}

// ---------------------------------------------------------------------------
// Weight convert + transpose: wt[n][k] = (f16) w[k][n], 512x512.
// ---------------------------------------------------------------------------
__global__ void wconvert_kernel(const float* __restrict__ w, _Float16* __restrict__ wt) {
  const int idx = blockIdx.x * 256 + threadIdx.x;   // 0 .. 262143
  const int n = idx >> 9, k = idx & 511;
  wt[(size_t)n * 512 + k] = (_Float16)w[(size_t)k * 512 + n];
}

// ---------------------------------------------------------------------------
// GEMM: C[b] = A[b] (4096x512 f16) @ Wt^T (Wt stored [n][k] f16) + bias.
// Grid (128 m-tiles of 32 rows, 2 batches), 256 threads = 8 waves.
// Wave w owns 32x64 output tile: rows [mt*32, +32), cols [w*64, +64).
// K loop software-pipelined with explicit ping-pong register buffers
// (unroll-by-2) so buffer rotation needs no v_mov copies: load k+1 into
// buf1, compute k from buf0, load k+2 into buf0, compute k+1 from buf1.
// __launch_bounds__(256, 4): 4 waves/SIMD -> 256-VGPR budget, no spills.
// Epilogue modes:
//   outF != 0 : f32 store with residual add (final projection)
//   tr   != 0 : f16 store transposed [512][4096] (for V)
//   else      : f16 store [4096][512]
// ---------------------------------------------------------------------------
__global__ void __launch_bounds__(256, 4)
gemm_kernel(const _Float16* __restrict__ A,
            const _Float16* __restrict__ Wt,
            const float* __restrict__ bias,
            const float* __restrict__ resid,
            _Float16* __restrict__ outH,
            float* __restrict__ outF,
            int tr) {
  const int b    = blockIdx.y;
  const int mt   = blockIdx.x;           // 32-row tile
  const int wave = threadIdx.x >> 5;
  const int lane = threadIdx.x & 31;

  const _Float16* A0 = A + ((size_t)b * 4096 + mt * 32) * 512;
  const _Float16* A1 = A0 + 16 * 512;
  const _Float16* Bw = Wt + (size_t)(wave * 64) * 512;   // 4 B tiles, stride 16*512

  v8f acc[2][4];
#pragma unroll
  for (int m = 0; m < 2; ++m)
#pragma unroll
    for (int j = 0; j < 4; ++j)
#pragma unroll
      for (int r = 0; r < 8; ++r) acc[m][j][r] = 0.f;

  // ping-pong fragment buffers
  v16h a0[2], a1[2], bf[2][4];

  // prologue: step 0 into buffer set 0
  a0[0] = load_frag_a(A0, 512);
  a1[0] = load_frag_a(A1, 512);
#pragma unroll
  for (int j = 0; j < 4; ++j) bf[0][j] = load_frag_b(Bw + (size_t)j * 16 * 512, 512);

#pragma unroll 1
  for (int i = 0; i < 7; ++i) {          // steps 0..13, two per iteration
    const int kt = i * 64;
    // prefetch upcoming lines into near caches (gfx1250 global_prefetch_b8)
    __builtin_prefetch(A0 + (size_t)(lane & 15) * 512 + kt + 192, 0, 3);
    __builtin_prefetch(Bw + (size_t)(lane & 15) * 512 + kt + 192, 0, 3);
    // load step kt+32 into set 1
    a0[1] = load_frag_a(A0 + kt + 32, 512);
    a1[1] = load_frag_a(A1 + kt + 32, 512);
#pragma unroll
    for (int j = 0; j < 4; ++j)
      bf[1][j] = load_frag_b(Bw + (size_t)j * 16 * 512 + kt + 32, 512);
    // compute step kt from set 0
#pragma unroll
    for (int j = 0; j < 4; ++j) {
      acc[0][j] = WMMA_F16(a0[0], bf[0][j], acc[0][j]);
      acc[1][j] = WMMA_F16(a1[0], bf[0][j], acc[1][j]);
    }
    // load step kt+64 into set 0
    a0[0] = load_frag_a(A0 + kt + 64, 512);
    a1[0] = load_frag_a(A1 + kt + 64, 512);
#pragma unroll
    for (int j = 0; j < 4; ++j)
      bf[0][j] = load_frag_b(Bw + (size_t)j * 16 * 512 + kt + 64, 512);
    // compute step kt+32 from set 1
#pragma unroll
    for (int j = 0; j < 4; ++j) {
      acc[0][j] = WMMA_F16(a0[1], bf[1][j], acc[0][j]);
      acc[1][j] = WMMA_F16(a1[1], bf[1][j], acc[1][j]);
    }
  }
  // tail: set 0 holds step 14 (kt=448); load step 15 (kt=480) into set 1
  a0[1] = load_frag_a(A0 + 480, 512);
  a1[1] = load_frag_a(A1 + 480, 512);
#pragma unroll
  for (int j = 0; j < 4; ++j)
    bf[1][j] = load_frag_b(Bw + (size_t)j * 16 * 512 + 480, 512);
#pragma unroll
  for (int j = 0; j < 4; ++j) {
    acc[0][j] = WMMA_F16(a0[0], bf[0][j], acc[0][j]);
    acc[1][j] = WMMA_F16(a1[0], bf[0][j], acc[1][j]);
  }
#pragma unroll
  for (int j = 0; j < 4; ++j) {
    acc[0][j] = WMMA_F16(a0[1], bf[1][j], acc[0][j]);
    acc[1][j] = WMMA_F16(a1[1], bf[1][j], acc[1][j]);
  }

  const int hi = lane >> 4, n0 = lane & 15;
#pragma unroll
  for (int m = 0; m < 2; ++m) {
#pragma unroll
    for (int j = 0; j < 4; ++j) {
      const int col  = wave * 64 + j * 16 + n0;
      const float bv = bias[col];
#pragma unroll
      for (int r = 0; r < 8; ++r) {
        const int row = mt * 32 + m * 16 + r + 8 * hi;
        const float v = acc[m][j][r] + bv;
        if (outF) {
          const size_t o = ((size_t)b * 4096 + row) * 512 + col;
          outF[o] = v + resid[o];
        } else if (tr) {
          outH[((size_t)b * 512 + col) * 4096 + row] = (_Float16)v;
        } else {
          outH[((size_t)b * 4096 + row) * 512 + col] = (_Float16)v;
        }
      }
    }
  }
}

// ---------------------------------------------------------------------------
// Flash attention: 16 queries per workgroup, online softmax over 64-key
// blocks. 8 waves: split-K halves for S (LDS reduce), then each wave owns
// 64 output channels with register-resident f32 accumulators.
// Q fragments for the wave's K-range are cached in registers across the
// entire key loop (loop invariant). Softmax is parallel across all 256
// threads (4 elements each + two LDS tree reductions).
// Q,K f16 [b][4096][512]; Vt f16 [b][512][4096]; O f16 [b][4096][512].
// ---------------------------------------------------------------------------
__global__ void __launch_bounds__(256, 4)
flash_kernel(const _Float16* __restrict__ Q,
             const _Float16* __restrict__ K,
             const _Float16* __restrict__ Vt,
             _Float16* __restrict__ O) {
  const int b    = blockIdx.y;
  const int mt   = blockIdx.x;
  const int wave = threadIdx.x >> 5;
  const int lane = threadIdx.x & 31;
  const int t    = threadIdx.x;

  const _Float16* Qb = Q  + ((size_t)b * 4096 + mt * 16) * 512;
  const _Float16* Kb = K  +  (size_t)b * 4096 * 512;
  const _Float16* Vb = Vt +  (size_t)b * 512 * 4096;

  __shared__ float    Sp[2][16][64];   // split-K partial scores (8 KB)
  __shared__ _Float16 P[16][64];       // softmax probabilities (2 KB)
  __shared__ float    red[16][16];     // row reductions (1 KB)
  __shared__ float    mrow[16], lrow[16], arow[16];

  if (t < 16) { mrow[t] = -1e30f; lrow[t] = 0.f; }

  const int tile = wave & 3;           // S column tile this wave computes
  const int kh   = wave >> 2;          // K half: 0 -> [0,256), 1 -> [256,512)

  // cache Q fragments for this wave's K range (loop invariant, 64 VGPRs)
  v16h qf[8];
#pragma unroll
  for (int j = 0; j < 8; ++j) qf[j] = load_frag_a(Qb + kh * 256 + j * 32, 512);

  v8f acc[4];
#pragma unroll
  for (int j = 0; j < 4; ++j)
#pragma unroll
    for (int r = 0; r < 8; ++r) acc[j][r] = 0.f;

  const float scale = 0.044194173824159216f;  // 512^-0.5
  const int   r16   = t & 15;                 // softmax row for this thread
  const int   cg    = t >> 4;                 // softmax column group (4 cols)

  for (int kb = 0; kb < 4096; kb += 64) {
    // --- Phase A: S[16x64] partials; only K fragments stream (double-buffered)
    {
      const _Float16* Kt = Kb + (size_t)(kb + tile * 16) * 512 + kh * 256;
      v8f s;
#pragma unroll
      for (int r = 0; r < 8; ++r) s[r] = 0.f;
      v16h kf = load_frag_b(Kt, 512);
#pragma unroll
      for (int j = 0; j < 7; ++j) {
        v16h kn = load_frag_b(Kt + (j + 1) * 32, 512);
        s = WMMA_F16(qf[j], kf, s);
        kf = kn;
      }
      s = WMMA_F16(qf[7], kf, s);
      const int hi = lane >> 4, n0 = lane & 15;
#pragma unroll
      for (int r = 0; r < 8; ++r) Sp[kh][r + 8 * hi][tile * 16 + n0] = s[r];
    }
    __syncthreads();

    // --- Phase B: parallel online softmax (thread -> row r16, cols cg*4..+3)
    float v4[4];
    float lm = -1e30f;
#pragma unroll
    for (int i = 0; i < 4; ++i) {
      v4[i] = (Sp[0][r16][cg * 4 + i] + Sp[1][r16][cg * 4 + i]) * scale;
      lm = fmaxf(lm, v4[i]);
    }
    red[r16][cg] = lm;
    __syncthreads();
    if (cg == 0) {
      float mx = mrow[r16];
#pragma unroll
      for (int i = 0; i < 16; ++i) mx = fmaxf(mx, red[r16][i]);
      arow[r16] = expf(mrow[r16] - mx);
      mrow[r16] = mx;
    }
    __syncthreads();
    {
      const float mx = mrow[r16];
      float ls = 0.f;
#pragma unroll
      for (int i = 0; i < 4; ++i) {
        const float p = expf(v4[i] - mx);
        P[r16][cg * 4 + i] = (_Float16)p;
        ls += p;
      }
      red[r16][cg] = ls;
    }
    __syncthreads();
    if (cg == 0) {
      float l = lrow[r16] * arow[r16];
#pragma unroll
      for (int i = 0; i < 16; ++i) l += red[r16][i];
      lrow[r16] = l;
    }

    // --- Phase C: rescale + O += P @ V  (wave owns channels [wave*64, +64))
    {
      const int hi = lane >> 4;
      float al[8];
#pragma unroll
      for (int r = 0; r < 8; ++r) al[r] = arow[r + 8 * hi];
#pragma unroll
      for (int j = 0; j < 4; ++j)
#pragma unroll
        for (int r = 0; r < 8; ++r) acc[j][r] *= al[r];

      const v16h pa0 = load_frag_a(&P[0][0], 64);
      const v16h pa1 = load_frag_a(&P[0][0] + 32, 64);
#pragma unroll
      for (int j = 0; j < 4; ++j) {
        const _Float16* Vp = Vb + (size_t)(wave * 64 + j * 16) * 4096 + kb;
        const v16h b0 = load_frag_b(Vp, 4096);
        const v16h b1 = load_frag_b(Vp + 32, 4096);
        acc[j] = WMMA_F16(pa0, b0, acc[j]);
        acc[j] = WMMA_F16(pa1, b1, acc[j]);
      }
    }
    __syncthreads();
  }

  // --- Epilogue: O = acc / l, store f16
  const int hi = lane >> 4, n0 = lane & 15;
  float li[8];
#pragma unroll
  for (int r = 0; r < 8; ++r) li[r] = 1.0f / lrow[r + 8 * hi];
  _Float16* Ob = O + ((size_t)b * 4096 + mt * 16) * 512;
#pragma unroll
  for (int j = 0; j < 4; ++j) {
    const int col = wave * 64 + j * 16 + n0;
#pragma unroll
    for (int r = 0; r < 8; ++r)
      Ob[(size_t)(r + 8 * hi) * 512 + col] = (_Float16)(acc[j][r] * li[r]);
  }
}

// ---------------------------------------------------------------------------
// Launch
// ---------------------------------------------------------------------------
extern "C" void kernel_launch(void* const* d_in, const int* in_sizes, int n_in,
                              void* d_out, int out_size, void* d_ws, size_t ws_size,
                              hipStream_t stream) {
  (void)in_sizes; (void)n_in; (void)out_size; (void)ws_size;
  const float* x     = (const float*)d_in[0];
  const float* gamma = (const float*)d_in[1];
  const float* beta  = (const float*)d_in[2];
  const float* wq    = (const float*)d_in[3];
  const float* bq    = (const float*)d_in[4];
  const float* wk    = (const float*)d_in[5];
  const float* bk    = (const float*)d_in[6];
  const float* wv    = (const float*)d_in[7];
  const float* bv    = (const float*)d_in[8];
  const float* wo    = (const float*)d_in[9];
  const float* bo    = (const float*)d_in[10];
  float* out = (float*)d_out;

  // workspace layout (f16 elements)
  _Float16* ws  = (_Float16*)d_ws;
  const size_t NC = (size_t)2 * 4096 * 512;   // 4,194,304
  const size_t WW = (size_t)512 * 512;        //   262,144
  _Float16* hn  = ws;
  _Float16* wqT = hn  + NC;
  _Float16* wkT = wqT + WW;
  _Float16* wvT = wkT + WW;
  _Float16* woT = wvT + WW;
  _Float16* Qf  = woT + WW;
  _Float16* Kf  = Qf  + NC;
  _Float16* Vtr = Kf  + NC;
  _Float16* Of  = Vtr + NC;

  hipLaunchKernelGGL(groupnorm_kernel, dim3(64), dim3(256), 0, stream, x, gamma, beta, hn);

  hipLaunchKernelGGL(wconvert_kernel, dim3(1024), dim3(256), 0, stream, wq, wqT);
  hipLaunchKernelGGL(wconvert_kernel, dim3(1024), dim3(256), 0, stream, wk, wkT);
  hipLaunchKernelGGL(wconvert_kernel, dim3(1024), dim3(256), 0, stream, wv, wvT);
  hipLaunchKernelGGL(wconvert_kernel, dim3(1024), dim3(256), 0, stream, wo, woT);

  const dim3 gGemm(128, 2);
  hipLaunchKernelGGL(gemm_kernel, gGemm, dim3(256), 0, stream, hn, wqT, bq, (const float*)nullptr, Qf,  (float*)nullptr, 0);
  hipLaunchKernelGGL(gemm_kernel, gGemm, dim3(256), 0, stream, hn, wkT, bk, (const float*)nullptr, Kf,  (float*)nullptr, 0);
  hipLaunchKernelGGL(gemm_kernel, gGemm, dim3(256), 0, stream, hn, wvT, bv, (const float*)nullptr, Vtr, (float*)nullptr, 1);

  hipLaunchKernelGGL(flash_kernel, dim3(256, 2), dim3(256), 0, stream, Qf, Kf, Vtr, Of);

  hipLaunchKernelGGL(gemm_kernel, gGemm, dim3(256), 0, stream, Of, woT, bo, x, (_Float16*)nullptr, out, 0);
}
